// AttnDecoderRNN_22041772163134
// MI455X (gfx1250) — matrix operand order repaired and checked by reading, not compiled, not run on hardware
//
#include <hip/hip_runtime.h>

typedef __bf16 bf16_t;
typedef __attribute__((ext_vector_type(16))) __bf16 v16bf;
typedef __attribute__((ext_vector_type(8)))  float  v8f;

#define BN   64      // batch
#define PP   196     // pixels (14*14)
#define DEC  2048    // encoder dim
#define AD   512     // attention/decoder/embed dim
#define VV   10000   // vocab
#define TMAX 29      // decode steps (T-1)

// ---------- helpers ----------
__device__ __forceinline__ bf16_t f2bf(float f) {
  union { float f; unsigned u; } in; in.f = f;
  unsigned r = (in.u + 0x7FFFu + ((in.u >> 16) & 1u)) >> 16;   // RNE
  union { unsigned short s; bf16_t b; } out; out.s = (unsigned short)r;
  return out.b;
}
__device__ __forceinline__ float sigf(float x) { return 1.f / (1.f + __expf(-x)); }

// A-fragment scatter index for element (m, k); frag layout [Mt][Kt][32][16]
// lane<16 : M=m%16, elems 0..7 = K 0..7 of tile, elems 8..15 = K 16..23
// lane>=16: M=m%16, elems 0..7 = K 8..15,        elems 8..15 = K 24..31
__device__ __forceinline__ size_t afrag_idx(int m, int k, int Kt) {
  int mt = m >> 4, kt = k >> 5, r = k & 31;
  int lane, e;
  if (r < 8)       { lane = m & 15;        e = r;          }
  else if (r < 16) { lane = 16 + (m & 15); e = r - 8;      }
  else if (r < 24) { lane = m & 15;        e = 8 + r - 16; }
  else             { lane = 16 + (m & 15); e = 8 + r - 24; }
  return (((size_t)(mt * Kt + kt) * 32) + (size_t)lane) * 16 + (size_t)e;
}

// ---------- pack fp32 activation [Mt*16, K] -> bf16 A-fragments ----------
__global__ void pack_a_kernel(const float* __restrict__ X, bf16_t* __restrict__ F,
                              int Mt, int Kt, int K) {
  int id = blockIdx.x * blockDim.x + threadIdx.x;
  int total = Mt * Kt * 32;
  if (id >= total) return;
  int lane = id & 31;
  int tile = id >> 5;
  int kt = tile % Kt, mt = tile / Kt;
  int m  = mt * 16 + (lane & 15);
  int k0 = kt * 32 + ((lane >> 4) ? 8 : 0);
  const float* src = X + (size_t)m * K + k0;
  bf16_t* dst = F + (size_t)id * 16;
#pragma unroll
  for (int e = 0; e < 8; ++e) {
    dst[e]     = f2bf(src[e]);
    dst[8 + e] = f2bf(src[16 + e]);
  }
}

// ---------- pack fp32 weight [K, N] row-major -> bf16 B-fragments [Kt][Nt][32][16] ----------
// lane<16 : N=nt*16+lane,    elems 0..7 = W[kt*32+0..7, N],  elems 8..15 = W[kt*32+16..23, N]
// lane>=16: N=nt*16+lane-16, elems 0..7 = W[kt*32+8..15, N], elems 8..15 = W[kt*32+24..31, N]
__global__ void pack_b_kernel(const float* __restrict__ W, bf16_t* __restrict__ F,
                              int Kt, int Nt, int N) {
  int id = blockIdx.x * blockDim.x + threadIdx.x;
  int total = Kt * Nt * 32;
  if (id >= total) return;
  int lane = id & 31;
  int tile = id >> 5;
  int nt = tile % Nt, kt = tile / Nt;
  int n = nt * 16 + (lane & 15);
  int kb = kt * 32 + ((lane >> 4) ? 8 : 0);
  bf16_t* dst = F + (size_t)id * 16;
#pragma unroll
  for (int e = 0; e < 16; ++e)
    dst[e] = f2bf(W[(size_t)(kb + e + (e >= 8 ? 8 : 0)) * N + n]);
}

// ---------- WMMA GEMM: one wave computes a 16 x (16*NTW) C strip ----------
// A fragment loaded once per kt, reused across NTW WMMAs.
// If caplen != null: rows with t >= caplen[m]-1 are zeroed.
template <int NTW>
__global__ void wmma_gemm_kernel(const bf16_t* __restrict__ Afrag,
                                 const bf16_t* __restrict__ Bfrag,
                                 const float* __restrict__ bias,
                                 float* __restrict__ C,
                                 int Mt, int Nt, int Kt,
                                 long long ldc, int accumulate,
                                 const int* __restrict__ caplen, int tstep) {
  int wave = (blockIdx.x * blockDim.x + threadIdx.x) >> 5;
  int lane = threadIdx.x & 31;
  int strips = Nt / NTW;
  if (wave >= Mt * strips) return;
  int mt = wave / strips, nt0 = (wave % strips) * NTW;

  v8f acc[NTW];
#pragma unroll
  for (int w2 = 0; w2 < NTW; ++w2) acc[w2] = v8f{};

  const bf16_t* ap = Afrag + ((size_t)mt * Kt) * 512 + (size_t)lane * 16;
  const bf16_t* bp = Bfrag + ((size_t)nt0) * 512 + (size_t)lane * 16;
  const size_t bstride = (size_t)Nt * 512;
  for (int kt = 0; kt < Kt; ++kt) {
    v16bf a = *reinterpret_cast<const v16bf*>(ap);
#pragma unroll
    for (int w2 = 0; w2 < NTW; ++w2) {
      v16bf b = *reinterpret_cast<const v16bf*>(bp + (size_t)w2 * 512);
      acc[w2] = __builtin_amdgcn_wmma_f32_16x16x32_bf16(false, a, false, b,
                                                        (short)0, acc[w2], false, false);
    }
    ap += 512;
    bp += bstride;
  }

  int nn_lane = lane & 15;
  int mbase = mt * 16 + ((lane >> 4) << 3);
#pragma unroll
  for (int w2 = 0; w2 < NTW; ++w2) {
    int n = (nt0 + w2) * 16 + nn_lane;
    float bv = bias ? bias[n] : 0.f;
#pragma unroll
    for (int r = 0; r < 8; ++r) {
      int m = mbase + r;
      float v = acc[w2][r] + bv;
      if (caplen && !(tstep < (caplen[m] - 1))) v = 0.f;
      size_t idx = (size_t)m * (size_t)ldc + (size_t)n;
      if (accumulate) C[idx] += v; else C[idx] = v;
    }
  }
}

// ---------- fused attention score + softmax (one block per batch row) ----------
// e[p] = relu(att1[b,p,:] + att2[b,:]) . w + fb ; alpha = softmax(e)
__global__ void att_softmax_kernel(const float* __restrict__ att1,
                                   const float* __restrict__ att2,
                                   const float* __restrict__ w,
                                   const float* __restrict__ fb,
                                   float* __restrict__ alpha,
                                   float* __restrict__ out_alphas,
                                   const int* __restrict__ caplen, int t) {
  __shared__ float es[PP];
  __shared__ float red[256];
  int b = blockIdx.x;
  int tid = threadIdx.x, lane = tid & 31, wv = tid >> 5;   // 8 waves
  const float* a2 = att2 + (size_t)b * AD;
  for (int p = wv; p < PP; p += 8) {
    const float* a1 = att1 + ((size_t)b * PP + p) * AD;
    float s = 0.f;
#pragma unroll 4
    for (int i = lane; i < AD; i += 32) {
      float v = a1[i] + a2[i];
      v = v > 0.f ? v : 0.f;
      s += v * w[i];
    }
    for (int o = 16; o; o >>= 1) s += __shfl_down(s, o, 32);
    if (lane == 0) es[p] = s + fb[0];
  }
  __syncthreads();
  float v = (tid < PP) ? es[tid] : -1e30f;
  red[tid] = v; __syncthreads();
  for (int s = 128; s; s >>= 1) { if (tid < s) red[tid] = fmaxf(red[tid], red[tid + s]); __syncthreads(); }
  float mx = red[0]; __syncthreads();
  float ex = (tid < PP) ? __expf(v - mx) : 0.f;
  red[tid] = ex; __syncthreads();
  for (int s = 128; s; s >>= 1) { if (tid < s) red[tid] += red[tid + s]; __syncthreads(); }
  float inv = 1.f / red[0];
  if (tid < PP) {
    float a = ex * inv;
    alpha[b * PP + tid] = a;
    int active = t < (caplen[b] - 1);
    out_alphas[(size_t)b * TMAX * PP + (size_t)t * PP + tid] = active ? a : 0.f;
  }
}

// ---------- awe[b,d] = sum_p alpha[b,p] * enc[b,p,d] ----------
__global__ void awe_kernel(const float* __restrict__ alpha, const float* __restrict__ enc,
                           float* __restrict__ awe) {
  int id = blockIdx.x * blockDim.x + threadIdx.x;
  if (id >= BN * DEC) return;
  int b = id >> 11, d = id & (DEC - 1);
  const float* ep = enc + (size_t)b * PP * DEC + d;
  const float* ap = alpha + b * PP;
  float s = 0.f;
  for (int p = 0; p < PP; ++p) s += ap[p] * ep[(size_t)p * DEC];
  awe[id] = s;
}

// ---------- x = concat(emb[caps[b,t]], sigmoid(gate_lin)*awe), written as bf16 A-frag ----------
__global__ void build_x_frag_kernel(const float* __restrict__ emb_tab,
                                    const int* __restrict__ caps, int t,
                                    const float* __restrict__ gate_lin,
                                    const float* __restrict__ awe,
                                    bf16_t* __restrict__ xfrag) {
  int id = blockIdx.x * blockDim.x + threadIdx.x;
  if (id >= BN * (AD + DEC)) return;
  int b = id / (AD + DEC), j = id % (AD + DEC);
  float v;
  if (j < AD) {
    v = emb_tab[(size_t)caps[b * 30 + t] * AD + j];
  } else {
    int j2 = j - AD;
    v = sigf(gate_lin[b * DEC + j2]) * awe[b * DEC + j2];
  }
  xfrag[afrag_idx(b, j, (AD + DEC) / 32)] = f2bf(v);
}

// ---------- LSTM cell: masked state update + direct A-frag emission ----------
__global__ void lstm_cell_kernel(const float* __restrict__ gates, float* __restrict__ h,
                                 float* __restrict__ c,
                                 bf16_t* __restrict__ hfrag, bf16_t* __restrict__ hnfrag,
                                 const int* __restrict__ caplen, int t) {
  int id = blockIdx.x * blockDim.x + threadIdx.x;
  if (id >= BN * AD) return;
  int b = id >> 9, j = id & (AD - 1);
  const float* g = gates + (size_t)b * 4 * AD;
  float ig = sigf(g[j]);
  float fg = sigf(g[AD + j]);
  float gg = tanhf(g[2 * AD + j]);
  float og = sigf(g[3 * AD + j]);
  float cn = fg * c[id] + ig * gg;
  float hn = og * tanhf(cn);
  int active = t < (caplen[b] - 1);
  float hcur = active ? hn : h[id];
  float ccur = active ? cn : c[id];
  h[id] = hcur;
  c[id] = ccur;
  size_t fi = afrag_idx(b, j, AD / 32);
  hfrag[fi]  = f2bf(hcur);   // carried (masked) h -> next-step GEMMs
  hnfrag[fi] = f2bf(hn);     // unmasked h_new -> fc GEMM
}

// ---------- mean over P ----------
__global__ void mean_enc_kernel(const float* __restrict__ enc, float* __restrict__ mean) {
  int id = blockIdx.x * blockDim.x + threadIdx.x;
  if (id >= BN * DEC) return;
  int b = id >> 11, d = id & (DEC - 1);
  const float* p = enc + (size_t)b * PP * DEC + d;
  float s = 0.f;
  for (int i = 0; i < PP; ++i) s += p[(size_t)i * DEC];
  mean[id] = s * (1.f / (float)PP);
}

// ---------- passthrough outputs (int bits) ----------
__global__ void copy_misc_kernel(const int* __restrict__ caps, const int* __restrict__ caplen,
                                 int* __restrict__ out_caps, int* __restrict__ out_dlen) {
  int id = blockIdx.x * blockDim.x + threadIdx.x;
  if (id < BN * 30) out_caps[id] = caps[id];
  if (id < BN) out_dlen[id] = caplen[id] - 1;
}

// =====================================================================
static inline int cdivi(long long a, long long b) { return (int)((a + b - 1) / b); }

template <int NTW>
static void launch_gemm(const bf16_t* A, const bf16_t* Bf, const float* bias, float* C,
                        int Mt, int Nt, int Kt, long long ldc, int accum,
                        const int* mk, int t, hipStream_t stream) {
  long long waves = (long long)Mt * (Nt / NTW);
  wmma_gemm_kernel<NTW><<<cdivi(waves * 32, 256), 256, 0, stream>>>(
      A, Bf, bias, C, Mt, Nt, Kt, ldc, accum, mk, t);
}

extern "C" void kernel_launch(void* const* d_in, const int* in_sizes, int n_in,
                              void* d_out, int out_size, void* d_ws, size_t ws_size,
                              hipStream_t stream) {
  (void)in_sizes; (void)n_in; (void)out_size; (void)ws_size;

  const float* enc        = (const float*)d_in[0];   // [64,196,2048]
  const int*   caps       = (const int*)  d_in[1];   // [64,30]
  const int*   caplen     = (const int*)  d_in[2];   // [64,1]
  const float* embedding  = (const float*)d_in[3];   // [10000,512]
  const float* enc_att_w  = (const float*)d_in[4];
  const float* enc_att_b  = (const float*)d_in[5];
  const float* dec_att_w  = (const float*)d_in[6];
  const float* dec_att_b  = (const float*)d_in[7];
  const float* full_att_w = (const float*)d_in[8];
  const float* full_att_b = (const float*)d_in[9];
  const float* init_h_w   = (const float*)d_in[10];
  const float* init_h_b   = (const float*)d_in[11];
  const float* init_c_w   = (const float*)d_in[12];
  const float* init_c_b   = (const float*)d_in[13];
  const float* f_beta_w   = (const float*)d_in[14];
  const float* f_beta_b   = (const float*)d_in[15];
  const float* w_ih       = (const float*)d_in[16];
  const float* w_hh       = (const float*)d_in[17];
  const float* b_ih       = (const float*)d_in[18];
  const float* b_hh       = (const float*)d_in[19];
  const float* fc_w       = (const float*)d_in[20];
  const float* fc_b       = (const float*)d_in[21];

  // output layout: predictions[64,29,10000] | captions[64,30] | dec_len[64] | alphas[64,29,196]
  float* out_pred   = (float*)d_out;
  int*   out_caps   = (int*)d_out + (long long)BN * TMAX * VV;
  int*   out_dlen   = out_caps + BN * 30;
  float* out_alphas = (float*)(out_dlen + BN);

  // ---- workspace bump allocator ----
  char* wsp = (char*)d_ws;
  size_t off = 0;
  auto alloc = [&](size_t bytes) -> void* {
    void* p = wsp + off;
    off = (off + bytes + 255) & ~(size_t)255;
    return p;
  };

  // bf16 B-fragments for all weights
  bf16_t* bf_dec   = (bf16_t*)alloc((size_t)AD * AD * 2);             // Kt16 Nt32
  bf16_t* bf_encat = (bf16_t*)alloc((size_t)DEC * AD * 2);            // Kt64 Nt32
  bf16_t* bf_ih0   = (bf16_t*)alloc((size_t)DEC * AD * 2);
  bf16_t* bf_ic0   = (bf16_t*)alloc((size_t)DEC * AD * 2);
  bf16_t* bf_fbeta = (bf16_t*)alloc((size_t)AD * DEC * 2);            // Kt16 Nt128
  bf16_t* bf_wih   = (bf16_t*)alloc((size_t)(AD + DEC) * 4 * AD * 2); // Kt80 Nt128
  bf16_t* bf_whh   = (bf16_t*)alloc((size_t)AD * 4 * AD * 2);         // Kt16 Nt128
  bf16_t* bf_fc    = (bf16_t*)alloc((size_t)AD * VV * 2);             // Kt16 Nt625

  // bf16 A-fragments
  bf16_t* af_enc   = (bf16_t*)alloc((size_t)BN * PP * DEC * 2);       // Mt784 Kt64
  bf16_t* af_mean  = (bf16_t*)alloc((size_t)BN * DEC * 2);            // Mt4 Kt64
  bf16_t* af_h     = (bf16_t*)alloc((size_t)BN * AD * 2);             // Mt4 Kt16
  bf16_t* af_hn    = (bf16_t*)alloc((size_t)BN * AD * 2);
  bf16_t* af_x     = (bf16_t*)alloc((size_t)BN * (AD + DEC) * 2);     // Mt4 Kt80

  // fp32 scratch
  float* att1     = (float*)alloc((size_t)BN * PP * AD * 4);          // [12544,512]
  float* mean     = (float*)alloc((size_t)BN * DEC * 4);
  float* h_buf    = (float*)alloc((size_t)BN * AD * 4);
  float* c_buf    = (float*)alloc((size_t)BN * AD * 4);
  float* att2     = (float*)alloc((size_t)BN * AD * 4);
  float* alpha    = (float*)alloc((size_t)BN * PP * 4);
  float* awe      = (float*)alloc((size_t)BN * DEC * 4);
  float* gate_lin = (float*)alloc((size_t)BN * DEC * 4);
  float* gates    = (float*)alloc((size_t)BN * 4 * AD * 4);

  const int BLK = 256;
  auto pb = [&](const float* W, bf16_t* F, int Kt, int Nt, int N) {
    pack_b_kernel<<<cdivi((long long)Kt * Nt * 32, BLK), BLK, 0, stream>>>(W, F, Kt, Nt, N);
  };
  auto pa = [&](const float* X, bf16_t* F, int Mt, int Kt, int K) {
    pack_a_kernel<<<cdivi((long long)Mt * Kt * 32, BLK), BLK, 0, stream>>>(X, F, Mt, Kt, K);
  };

  // ---------- setup ----------
  pb(dec_att_w, bf_dec,   16,  32, AD);
  pb(enc_att_w, bf_encat, 64,  32, AD);
  pb(init_h_w,  bf_ih0,   64,  32, AD);
  pb(init_c_w,  bf_ic0,   64,  32, AD);
  pb(f_beta_w,  bf_fbeta, 16, 128, DEC);
  pb(w_ih,      bf_wih,   80, 128, 4 * AD);
  pb(w_hh,      bf_whh,   16, 128, 4 * AD);
  pb(fc_w,      bf_fc,    16, 625, VV);

  copy_misc_kernel<<<cdivi(BN * 30, BLK), BLK, 0, stream>>>(caps, caplen, out_caps, out_dlen);

  mean_enc_kernel<<<cdivi(BN * DEC, BLK), BLK, 0, stream>>>(enc, mean);
  pa(mean, af_mean, 4, 64, DEC);
  launch_gemm<4>(af_mean, bf_ih0, init_h_b, h_buf, 4, 32, 64, AD, 0, nullptr, 0, stream);
  launch_gemm<4>(af_mean, bf_ic0, init_c_b, c_buf, 4, 32, 64, AD, 0, nullptr, 0, stream);

  pa(enc, af_enc, (BN * PP) / 16, 64, DEC);                           // Mt=784
  launch_gemm<4>(af_enc, bf_encat, enc_att_b, att1, (BN * PP) / 16, 32, 64, AD, 0,
                 nullptr, 0, stream);

  pa(h_buf, af_h, 4, 16, AD);

  // ---------- decode loop (static 29 steps; per-row masking == batch truncation) ----------
  for (int t = 0; t < TMAX; ++t) {
    launch_gemm<4>(af_h, bf_dec, dec_att_b, att2, 4, 32, 16, AD, 0, nullptr, 0, stream);
    att_softmax_kernel<<<BN, 256, 0, stream>>>(att1, att2, full_att_w, full_att_b,
                                               alpha, out_alphas, caplen, t);
    awe_kernel<<<cdivi(BN * DEC, BLK), BLK, 0, stream>>>(alpha, enc, awe);
    launch_gemm<4>(af_h, bf_fbeta, f_beta_b, gate_lin, 4, 128, 16, DEC, 0, nullptr, 0, stream);
    build_x_frag_kernel<<<cdivi((long long)BN * (AD + DEC), BLK), BLK, 0, stream>>>(
        embedding, caps, t, gate_lin, awe, af_x);
    launch_gemm<4>(af_x, bf_wih, b_ih, gates, 4, 128, 80, 4 * AD, 0, nullptr, 0, stream);
    launch_gemm<4>(af_h, bf_whh, b_hh, gates, 4, 128, 16, 4 * AD, 1, nullptr, 0, stream);
    lstm_cell_kernel<<<cdivi(BN * AD, BLK), BLK, 0, stream>>>(
        gates, h_buf, c_buf, af_h, af_hn, caplen, t);
    launch_gemm<5>(af_hn, bf_fc, fc_b, out_pred + (size_t)t * VV, 4, 625, 16,
                   (long long)TMAX * VV, 0, caplen, t, stream);      // masked rows -> 0
  }
}
